// MultiheadSelfAttention_5033701670935
// MI455X (gfx1250) — matrix operand order repaired and checked by reading, not compiled, not run on hardware
//
#include <hip/hip_runtime.h>

// ---------------------------------------------------------------------------
// MI455X (gfx1250) multihead self-attention, fp32 in/out, bf16 WMMA compute.
//  Stage 1: fused QKV projection GEMMs  -> Q,K bf16 [B,H,T,hd]; V bf16 [B,H,hd,T]
//  Stage 2: causal flash attention      -> out fp32 [B,T,D]
// Matrix math: v_wmma_f32_16x16x32_bf16.  K/V tiles staged to LDS via the
// Tensor Data Mover (tensor_load_to_lds + s_wait_tensorcnt) when available.
// ---------------------------------------------------------------------------

typedef unsigned int   u32;
typedef unsigned short u16;
typedef unsigned long long u64;
typedef __attribute__((ext_vector_type(16))) __bf16 v16bf;
typedef __attribute__((ext_vector_type(8)))  float  v8f;

union Frag16 { v16bf v; u32 u[8]; };

__device__ __forceinline__ u16 f2bf(float f) {       // f32 -> bf16, RNE
  u32 u = __float_as_uint(f);
  u = (u + 0x7FFFu + ((u >> 16) & 1u)) >> 16;
  return (u16)u;
}

__device__ __forceinline__ u32 pack_bf16(float lo, float hi) {
#if defined(__HIP_DEVICE_COMPILE__) && __has_builtin(__builtin_amdgcn_cvt_pk_bf16_f32)
  auto pk = __builtin_amdgcn_cvt_pk_bf16_f32(lo, hi);   // v_cvt_pk_bf16_f32
  u32 r;
  __builtin_memcpy(&r, &pk, sizeof(r));
  return r;
#else
  return (u32)f2bf(lo) | ((u32)f2bf(hi) << 16);
#endif
}

#define B_   2
#define T_   2048
#define D_   1024
#define H_   16
#define HD_  64
#define QKV_ELEMS (B_ * H_ * T_ * HD_)   // 4194304 bf16 elems per tensor

// ---------------------------------------------------------------------------
// Tensor Data Mover: 2-D tile load, bf16 (data_size=2B).  clang-23 6-arg form:
//   (uint32x4 g0, int32x8 g1, int32x4 g2, int32x4 g3, int32x8 pad, i32 cpol)
// ---------------------------------------------------------------------------
#if defined(__HIP_DEVICE_COMPILE__) && __has_builtin(__builtin_amdgcn_tensor_load_to_lds)
#define USE_TDM 1
typedef __attribute__((ext_vector_type(4))) unsigned int u32x4;
typedef __attribute__((ext_vector_type(8))) int          i32x8;
typedef __attribute__((ext_vector_type(4))) int          i32x4;

__device__ __forceinline__ void tdm_load_2d_bf16(u32 lds_addr, const void* gptr,
                                                 u32 dim0, u32 dim1, u32 stride0,
                                                 u32 tile0, u32 tile1) {
  const u64 ga = (u64)gptr;
  u32x4 g0;
  g0[0] = 1u;                                        // count=1, user descriptor
  g0[1] = lds_addr;                                  // lds_addr [63:32]
  g0[2] = (u32)ga;                                   // global_addr lo
  g0[3] = (u32)((ga >> 32) & 0x01FFFFFFu) | (2u << 30);  // addr[56:32] | type=2
  i32x8 g1;
  g1[0] = (int)(1u << 16);                           // wg_mask=0, data_size=1 (2B)
  g1[1] = (int)((dim0 & 0xFFFFu) << 16);             // tensor_dim0[15:0]
  g1[2] = (int)(((dim0 >> 16) & 0xFFFFu) | ((dim1 & 0xFFFFu) << 16));
  g1[3] = (int)(((dim1 >> 16) & 0xFFFFu) | ((tile0 & 0xFFFFu) << 16));
  g1[4] = (int)(tile1 & 0xFFFFu);                    // tile_dim1 (tile_dim2=0)
  g1[5] = (int)stride0;                              // tensor_dim0_stride lo
  g1[6] = 0;
  g1[7] = 0;
  i32x4 z4 = {0, 0, 0, 0};                           // dims 2..4 unused
  i32x8 z8 = {0, 0, 0, 0, 0, 0, 0, 0};
  __builtin_amdgcn_tensor_load_to_lds(g0, g1, z4, z4, z8, 0);
}
#endif

// ---------------------------------------------------------------------------
// Kernel 1: y = x @ W^T for W in {Wq,Wk,Wv}.  Q,K -> [B,H,T,hd]; V -> [B,H,hd,T].
// grid (M/64, H, 3), block 128 (4 waves); wave computes a 16x64 strip.
// ---------------------------------------------------------------------------
__global__ __launch_bounds__(128)
void qkv_proj_kernel(const float* __restrict__ x,
                     const float* __restrict__ Wq,
                     const float* __restrict__ Wk,
                     const float* __restrict__ Wv,
                     u16* __restrict__ ws)
{
  __shared__ u16 lds_a[64 * 32];   // x tile
  __shared__ u16 lds_b[64 * 32];   // W tile

  const int tid  = threadIdx.x;
  const int wave = tid >> 5;
  const int lane = tid & 31;
  const int m16  = lane & 15;
  const int half = lane >> 4;

  const int mtile = blockIdx.x;
  const int head  = blockIdx.y;
  const int which = blockIdx.z;              // 0=Q 1=K 2=V

  const float* W = (which == 0) ? Wq : (which == 1) ? Wk : Wv;
  u16* out = ws + (size_t)which * QKV_ELEMS;

  const int mbase = mtile * 64;
  const int nbase = head * HD_;

  v8f acc[4] = {};

  for (int k0 = 0; k0 < D_; k0 += 32) {
    __syncthreads();
    // cooperative stage: float4 loads + packed bf16 conversion (16 elem/thr)
#pragma unroll
    for (int i = 0; i < 4; ++i) {
      int e = tid + i * 128;          // 512 float4s per matrix
      int r = e >> 3, c4 = (e & 7) * 4;
      float4 xv = *(const float4*)&x[(size_t)(mbase + r) * D_ + k0 + c4];
      *(u32*)&lds_a[r * 32 + c4]     = pack_bf16(xv.x, xv.y);
      *(u32*)&lds_a[r * 32 + c4 + 2] = pack_bf16(xv.z, xv.w);
      float4 wv = *(const float4*)&W[(size_t)(nbase + r) * D_ + k0 + c4];
      *(u32*)&lds_b[r * 32 + c4]     = pack_bf16(wv.x, wv.y);
      *(u32*)&lds_b[r * 32 + c4 + 2] = pack_bf16(wv.z, wv.w);
    }
    // prefetch next K-step tiles into cache (global_prefetch_b8)
    if (k0 + 32 < D_) {
      int r = tid >> 1;
      __builtin_prefetch(&x[(size_t)(mbase + r) * D_ + k0 + 32 + (tid & 1) * 16], 0, 3);
      __builtin_prefetch(&W[(size_t)(nbase + r) * D_ + k0 + 32 + (tid & 1) * 16], 0, 3);
    }
    __syncthreads();

    // A fragment: 16x32 bf16 (VGPR0-3 -> K=8h+2v, VGPR4-7 -> K=16+8h+2v)
    Frag16 a;
    const int arow = (wave * 16 + m16) * 32;
#pragma unroll
    for (int v = 0; v < 4; ++v) {
      a.u[v]     = *(const u32*)&lds_a[arow + 8 * half + 2 * v];
      a.u[4 + v] = *(const u32*)&lds_a[arow + 16 + 8 * half + 2 * v];
    }
    // batch all four B fragments first, then issue WMMAs back-to-back
    Frag16 b[4];
#pragma unroll
    for (int nt = 0; nt < 4; ++nt) {
      const int brow = (nt * 16 + m16) * 32;
#pragma unroll
      for (int v = 0; v < 8; ++v)
        b[nt].u[v] = *(const u32*)&lds_b[brow + ((v & 4) << 2) + 8 * half + 2 * (v & 3)];
    }
#pragma unroll
    for (int nt = 0; nt < 4; ++nt)
      acc[nt] = __builtin_amdgcn_wmma_f32_16x16x32_bf16(
          false, a.v, false, b[nt].v, (short)0, acc[nt], false, false);
  }

  // Epilogue.  C layout: lane col n = m16, rows v + 8*half (t-consecutive).
  if (which == 2) {
    // V transposed: [B,H,hd,T] -> 8 consecutive t per lane -> one 16B store
#pragma unroll
    for (int nt = 0; nt < 4; ++nt) {
      int m0 = mbase + wave * 16 + 8 * half;
      int bb = m0 >> 11, t0 = m0 & (T_ - 1);
      int d  = nt * 16 + m16;
      uint4 pk;
      pk.x = pack_bf16(acc[nt][0], acc[nt][1]);
      pk.y = pack_bf16(acc[nt][2], acc[nt][3]);
      pk.z = pack_bf16(acc[nt][4], acc[nt][5]);
      pk.w = pack_bf16(acc[nt][6], acc[nt][7]);
      *(uint4*)&out[((size_t)(bb * H_ + head) * HD_ + d) * T_ + t0] = pk;
    }
  } else {
    // Q,K row-major: [B,H,T,hd]
#pragma unroll
    for (int nt = 0; nt < 4; ++nt) {
#pragma unroll
      for (int v = 0; v < 8; ++v) {
        int m = mbase + wave * 16 + v + 8 * half;
        int bb = m >> 11, t = m & (T_ - 1);
        int d = nt * 16 + m16;
        out[((size_t)(bb * H_ + head) * T_ + t) * HD_ + d] = f2bf(acc[nt][v]);
      }
    }
  }
}

// ---------------------------------------------------------------------------
// Kernel 2: causal flash attention.  grid (T/64, B*H), block 128 (4 waves).
// ---------------------------------------------------------------------------
__global__ __launch_bounds__(128)
void attn_kernel(const u16* __restrict__ qws,
                 const u16* __restrict__ kws,
                 const u16* __restrict__ vws,
                 float* __restrict__ outp)
{
  __shared__ u16 lds_k [64 * 64];      // K tile, [key][hd]
  __shared__ u16 lds_vt[64 * 64];      // V tile, [hd][key]  (global is [hd][T])
  __shared__ u16 lds_p [4 * 16 * 64];  // per-wave P (16 rows x 64 keys)

  const int tid  = threadIdx.x;
  const int wave = tid >> 5;
  const int lane = tid & 31;
  const int m16  = lane & 15;
  const int half = lane >> 4;

  const int qtile = blockIdx.x;
  const int bh    = blockIdx.y;
  const int head  = bh & (H_ - 1);
  const int bidx  = bh >> 4;

  const int   qbase = qtile * 64;
  const float scl   = 0.125f;            // 1/sqrt(hd)
  const float NEGI  = -__builtin_inff();

  // Q fragments (A-layout), 2 hd-halves, loaded once from [B,H,T,hd]
  Frag16 qf[2];
  {
    const u16* qrow =
        qws + ((size_t)bh * T_ + (qbase + wave * 16 + m16)) * HD_;
#pragma unroll
    for (int kh = 0; kh < 2; ++kh) {
#pragma unroll
      for (int v = 0; v < 4; ++v) {
        qf[kh].u[v]     = *(const u32*)&qrow[kh * 32 + 8 * half + 2 * v];
        qf[kh].u[4 + v] = *(const u32*)&qrow[kh * 32 + 16 + 8 * half + 2 * v];
      }
    }
  }

  v8f   acc[4] = {};
  float m_run[8], l_run[8];
#pragma unroll
  for (int v = 0; v < 8; ++v) { m_run[v] = NEGI; l_run[v] = 0.0f; }

  for (int kt = 0; kt <= qtile; ++kt) {
    const int kbase = kt * 64;
    __syncthreads();                      // previous iteration's LDS reads done

#ifdef USE_TDM
    // Tensor Data Mover: one wave issues both tile DMAs; TDM ignores EXEC.
    if (wave == 0) {
      tdm_load_2d_bf16((u32)(size_t)&lds_k[0],
                       kws + ((size_t)bh * T_ + kbase) * HD_,
                       /*dim0=*/HD_, /*dim1=*/T_, /*stride0=*/HD_,
                       /*tile0=*/64, /*tile1=*/64);
      tdm_load_2d_bf16((u32)(size_t)&lds_vt[0],
                       vws + (size_t)bh * HD_ * T_ + kbase,
                       /*dim0=*/T_, /*dim1=*/HD_, /*stride0=*/T_,
                       /*tile0=*/64, /*tile1=*/64);
      __builtin_amdgcn_s_wait_tensorcnt(0);
    }
#else
    // manual cooperative staging: both tiles are straight copies now
#pragma unroll
    for (int i = 0; i < 16; ++i) {
      int e = tid + i * 128;
      int r = e >> 5, c2 = (e & 31) * 2;
      *(u32*)&lds_k[r * 64 + c2] =
          *(const u32*)&kws[((size_t)bh * T_ + kbase + r) * HD_ + c2];
      *(u32*)&lds_vt[r * 64 + c2] =
          *(const u32*)&vws[((size_t)bh * HD_ + r) * T_ + kbase + c2];
    }
#endif
    __syncthreads();

    // S = Q @ K^T : per key-col tile load both hd-half B frags, then 2 WMMAs
    v8f s[4];
#pragma unroll
    for (int nt = 0; nt < 4; ++nt) {
      Frag16 bk[2];
      const int krow = (nt * 16 + m16) * 64;
#pragma unroll
      for (int kh = 0; kh < 2; ++kh)
#pragma unroll
        for (int v = 0; v < 8; ++v)
          bk[kh].u[v] = *(const u32*)&lds_k[krow + kh * 32 + ((v & 4) << 2) + 8 * half + 2 * (v & 3)];
      v8f c = {};
      c = __builtin_amdgcn_wmma_f32_16x16x32_bf16(
          false, qf[0].v, false, bk[0].v, (short)0, c, false, false);
      c = __builtin_amdgcn_wmma_f32_16x16x32_bf16(
          false, qf[1].v, false, bk[1].v, (short)0, c, false, false);
      s[nt] = c;
    }

    // causal mask (only fires on the diagonal tile)
#pragma unroll
    for (int nt = 0; nt < 4; ++nt) {
      const int key = kbase + nt * 16 + m16;
#pragma unroll
      for (int v = 0; v < 8; ++v) {
        const int qr = qbase + wave * 16 + v + 8 * half;
        if (key > qr) s[nt][v] = NEGI;
      }
    }

    // online softmax (row lives across 16 lanes of a half)
    float mloc[8], mnew[8], sc[8], lloc[8];
#pragma unroll
    for (int v = 0; v < 8; ++v)
      mloc[v] = fmaxf(fmaxf(s[0][v], s[1][v]), fmaxf(s[2][v], s[3][v]));
#pragma unroll
    for (int off = 1; off < 16; off <<= 1)
#pragma unroll
      for (int v = 0; v < 8; ++v)
        mloc[v] = fmaxf(mloc[v], __shfl_xor(mloc[v], off, 16));
#pragma unroll
    for (int v = 0; v < 8; ++v) {
      mnew[v] = fmaxf(m_run[v], mloc[v]);
      sc[v]   = __expf((m_run[v] - mnew[v]) * scl);
      lloc[v] = 0.0f;
    }

    // P = exp((S - m) * scl); bf16 into per-wave LDS (C -> A relayout)
    u16* prow = &lds_p[wave * (16 * 64)];
#pragma unroll
    for (int nt = 0; nt < 4; ++nt) {
#pragma unroll
      for (int v = 0; v < 8; ++v) {
        float p = __expf((s[nt][v] - mnew[v]) * scl);
        lloc[v] += p;
        prow[(v + 8 * half) * 64 + nt * 16 + m16] = f2bf(p);
      }
    }
#pragma unroll
    for (int off = 1; off < 16; off <<= 1)
#pragma unroll
      for (int v = 0; v < 8; ++v)
        lloc[v] += __shfl_xor(lloc[v], off, 16);
#pragma unroll
    for (int v = 0; v < 8; ++v) {
      l_run[v] = l_run[v] * sc[v] + lloc[v];
      m_run[v] = mnew[v];
    }
#pragma unroll
    for (int nt = 0; nt < 4; ++nt)
#pragma unroll
      for (int v = 0; v < 8; ++v)
        acc[nt][v] *= sc[v];

    __syncthreads();   // P visible; keeps LDS phases uniform

    // O += P @ V : per key-half, batch 4 V B-frags then 4 WMMAs
#pragma unroll
    for (int kh = 0; kh < 2; ++kh) {
      Frag16 a;
      const u16* pr = &lds_p[wave * (16 * 64) + m16 * 64 + kh * 32];
#pragma unroll
      for (int v = 0; v < 4; ++v) {
        a.u[v]     = *(const u32*)&pr[8 * half + 2 * v];
        a.u[4 + v] = *(const u32*)&pr[16 + 8 * half + 2 * v];
      }
      Frag16 bv[4];
#pragma unroll
      for (int nt = 0; nt < 4; ++nt) {
        const int vrow = (nt * 16 + m16) * 64 + kh * 32;
#pragma unroll
        for (int v = 0; v < 8; ++v)
          bv[nt].u[v] = *(const u32*)&lds_vt[vrow + ((v & 4) << 2) + 8 * half + 2 * (v & 3)];
      }
#pragma unroll
      for (int nt = 0; nt < 4; ++nt)
        acc[nt] = __builtin_amdgcn_wmma_f32_16x16x32_bf16(
            false, a.v, false, bv[nt].v, (short)0, acc[nt], false, false);
    }
  }

  // finalize: out[b][q][head*64+d] = acc / l   (fp32)
#pragma unroll
  for (int nt = 0; nt < 4; ++nt) {
#pragma unroll
    for (int v = 0; v < 8; ++v) {
      const int q = qbase + wave * 16 + v + 8 * half;
      const int d = nt * 16 + m16;
      outp[((size_t)bidx * T_ + q) * D_ + head * HD_ + d] =
          acc[nt][v] / l_run[v];
    }
  }
}

// ---------------------------------------------------------------------------
extern "C" void kernel_launch(void* const* d_in, const int* in_sizes, int n_in,
                              void* d_out, int out_size, void* d_ws,
                              size_t ws_size, hipStream_t stream) {
  (void)in_sizes; (void)n_in; (void)out_size; (void)ws_size;
  const float* x  = (const float*)d_in[0];
  const float* Wq = (const float*)d_in[1];
  const float* Wk = (const float*)d_in[2];
  const float* Wv = (const float*)d_in[3];
  u16* ws = (u16*)d_ws;                      // Q | K | V  (bf16)

  qkv_proj_kernel<<<dim3(64, 16, 3), 128, 0, stream>>>(x, Wq, Wk, Wv, ws);
  attn_kernel<<<dim3(T_ / 64, B_ * H_), 128, 0, stream>>>(
      ws, ws + QKV_ELEMS, ws + 2 * (size_t)QKV_ELEMS, (float*)d_out);
}
